// LocalCrossAttention_90288802497335
// MI455X (gfx1250) — matrix operand order repaired and checked by reading, not compiled, not run on hardware
//
#include <hip/hip_runtime.h>

#define D_EMBED 1024
#define N_TOK   4096

typedef __attribute__((ext_vector_type(16))) _Float16 v16h;
typedef __attribute__((ext_vector_type(8)))  _Float16 v8h;
typedef __attribute__((ext_vector_type(4)))  _Float16 v4h;
typedef __attribute__((ext_vector_type(8)))  float    v8f;

// ---------------- WMMA fragment loaders (ISA 7.12.2 layouts) ----------------
// A 16x32 f16: lane L (L<16): m=L, K={0..7,16..23}; lane L+16: m=L, K={8..15,24..31}
__device__ __forceinline__ v16h a_frag_f16(const _Float16* row, int k0, int half) {
    const v8h* p0 = (const v8h*)(row + k0 + half * 8);
    const v8h* p1 = (const v8h*)(row + k0 + 16 + half * 8);
    v8h lo = *p0, hi = *p1;
    v16h r;
#pragma unroll
    for (int i = 0; i < 8; ++i) { r[i] = lo[i]; r[i + 8] = hi[i]; }
    return r;
}
// B 32x16 f16: lane L (L<16): n=L, K={0..7,16..23}; lane L+16: n=L, K={8..15,24..31}
__device__ __forceinline__ v16h b_frag_f16(const _Float16* row, int k0, int half) {
    const v8h* p0 = (const v8h*)(row + k0 + half * 8);
    const v8h* p1 = (const v8h*)(row + k0 + 16 + half * 8);
    v8h lo = *p0, hi = *p1;
    v16h r;
#pragma unroll
    for (int i = 0; i < 8; ++i) { r[i] = lo[i]; r[i + 8] = hi[i]; }
    return r;
}

// ---------------- Kernel 0: bulk fp32 -> f16 conversion ---------------------
__global__ __launch_bounds__(256) void cvt_kernel(const float* __restrict__ src,
                                                  _Float16* __restrict__ dst, int n4) {
    const int i = blockIdx.x * blockDim.x + threadIdx.x;
    if (i < n4) {
        const float4 v = ((const float4*)src)[i];
        v4h h;
        h[0] = (_Float16)v.x; h[1] = (_Float16)v.y;
        h[2] = (_Float16)v.z; h[3] = (_Float16)v.w;
        ((v4h*)dst)[i] = h;
    }
}

// ---------------- Kernel 1: Y = X @ W^T + b (f16 in, f16 out) ---------------
// block = 256 thr (8 waves) covers 16 rows x 512 cols; each wave owns 4 tiles
// (4 independent WMMA accumulator chains to cover XDL latency).
__global__ __launch_bounds__(256) void proj_kernel(
    const _Float16* __restrict__ Xh, const _Float16* __restrict__ Wh,
    const float* __restrict__ bias, _Float16* __restrict__ out, int transposed) {
    extern __shared__ __align__(16) char smem[];
    _Float16* sX = (_Float16*)smem;                        // 16 * 1024 f16 = 32 KB
    const int rbase = blockIdx.x * 16;
    const int wave  = threadIdx.x >> 5;
    const int lane  = threadIdx.x & 31;
    const int col = lane & 15, half = lane >> 4;

    { // stage X strip (pure 32 KB copy, already f16)
        const uint4* src = (const uint4*)(Xh + (size_t)rbase * D_EMBED);
        uint4* dst = (uint4*)sX;
        for (int i = threadIdx.x; i < 16 * D_EMBED / 8; i += blockDim.x) dst[i] = src[i];
    }
    __syncthreads();

    const int dbase = blockIdx.y * 512 + wave * 64;
    const _Float16* aRow = sX + col * D_EMBED;
    const _Float16* bRow[4];
#pragma unroll
    for (int t = 0; t < 4; ++t) bRow[t] = Wh + (size_t)(dbase + t * 16 + col) * D_EMBED;

    v8f acc[4] = {};
#pragma unroll 2
    for (int kk = 0; kk < D_EMBED; kk += 32) {
        v16h a = a_frag_f16(aRow, kk, half);
#pragma unroll
        for (int t = 0; t < 4; ++t) {
            v16h b = b_frag_f16(bRow[t], kk, half);
            acc[t] = __builtin_amdgcn_wmma_f32_16x16x32_f16(false, a, false, b,
                                                            (short)0, acc[t], false, false);
        }
    }
#pragma unroll
    for (int t = 0; t < 4; ++t) {
        const int d  = dbase + t * 16 + col;
        const float bv = bias[d];
        if (!transposed) {
#pragma unroll
            for (int r = 0; r < 8; ++r) {
                const int m = r + half * 8;
                out[(size_t)(rbase + m) * D_EMBED + d] = (_Float16)(acc[t][r] + bv);
            }
        } else {
            v8h pk;                                 // 8 consecutive n -> one b128 store
#pragma unroll
            for (int r = 0; r < 8; ++r) pk[r] = (_Float16)(acc[t][r] + bv);
            *(v8h*)(out + (size_t)d * N_TOK + rbase + half * 8) = pk;
        }
    }
}

// ---------------- Kernel 2: fused scores + softmax + context ----------------
// block = 256 thr (8 waves) owns 16 query rows; full 16x4096 score strip lives
// in LDS (f16, 128 KB) + Q strip (32 KB) -> single-pass exact softmax.
__global__ __launch_bounds__(256) void attn_kernel(
    const _Float16* __restrict__ q2, const _Float16* __restrict__ k1,
    const _Float16* __restrict__ v1t,
    const _Float16* __restrict__ q1, const _Float16* __restrict__ k2,
    const _Float16* __restrict__ v2t, float* __restrict__ out) {
    extern __shared__ __align__(16) char smem[];
    _Float16* sQ = (_Float16*)smem;                                    // 16*1024
    _Float16* sS = (_Float16*)(smem + 16 * D_EMBED * 2);               // 16*4096
    float*    sR = (float*)(smem + 16 * D_EMBED * 2 + 16 * N_TOK * 2); // 16

    const _Float16 *Q, *K, *V;
    float* O;
    if (blockIdx.y == 0) { Q = q2; K = k1; V = v1t; O = out; }                           // context1
    else                 { Q = q1; K = k2; V = v2t; O = out + (size_t)N_TOK * D_EMBED; } // context2

    const int rbase = blockIdx.x * 16;
    const int wave  = threadIdx.x >> 5;
    const int lane  = threadIdx.x & 31;
    const int col = lane & 15, half = lane >> 4;

    { // stage Q strip (contiguous 32 KB copy)
        const uint4* src = (const uint4*)(Q + (size_t)rbase * D_EMBED);
        uint4* dst = (uint4*)sQ;
        for (int i = threadIdx.x; i < 16 * D_EMBED / 8; i += blockDim.x) dst[i] = src[i];
    }
    __syncthreads();

    // ---- phase 1: S = scale * Q K^T; wave w covers keys [512w, 512w+512),
    //      four 16-key tiles (4 independent WMMA chains) per A-fragment load.
    const _Float16* aRow = sQ + col * D_EMBED;
    for (int nt = 0; nt < 8; ++nt) {
        const int nbase = wave * 512 + nt * 64;
        const _Float16* bRow[4];
#pragma unroll
        for (int t = 0; t < 4; ++t) bRow[t] = K + (size_t)(nbase + t * 16 + col) * D_EMBED;
        v8f acc[4] = {};
#pragma unroll 2
        for (int kk = 0; kk < D_EMBED; kk += 32) {
            v16h a = a_frag_f16(aRow, kk, half);
#pragma unroll
            for (int t = 0; t < 4; ++t) {
                v16h b = b_frag_f16(bRow[t], kk, half);
                acc[t] = __builtin_amdgcn_wmma_f32_16x16x32_f16(false, a, false, b,
                                                                (short)0, acc[t], false, false);
            }
        }
#pragma unroll
        for (int t = 0; t < 4; ++t)
#pragma unroll
            for (int r = 0; r < 8; ++r) {
                const int m = r + half * 8;
                sS[m * N_TOK + nbase + t * 16 + col] = (_Float16)(acc[t][r] * 0.03125f);
            }
    }
    __syncthreads();

    // ---- phase 2: exact softmax; wave w owns rows 2w, 2w+1 (16B LDS vectors)
#pragma unroll
    for (int rr = 0; rr < 2; ++rr) {
        const int m = wave * 2 + rr;
        v8h* srow = (v8h*)(sS + m * N_TOK);     // 512 x v8h per row
        float mx = -3.0e38f;
        for (int c = lane; c < N_TOK / 8; c += 32) {
            v8h v = srow[c];
#pragma unroll
            for (int i = 0; i < 8; ++i) mx = fmaxf(mx, (float)v[i]);
        }
#pragma unroll
        for (int off = 16; off > 0; off >>= 1) mx = fmaxf(mx, __shfl_xor(mx, off, 32));
        float sum = 0.f;
        for (int c = lane; c < N_TOK / 8; c += 32) {
            v8h v = srow[c], pv;
#pragma unroll
            for (int i = 0; i < 8; ++i) {
                float p = __expf((float)v[i] - mx);
                sum += p;
                pv[i] = (_Float16)p;
            }
            srow[c] = pv;                       // P in place, already A-fragment layout
        }
#pragma unroll
        for (int off = 16; off > 0; off >>= 1) sum += __shfl_xor(sum, off, 32);
        if (lane == 0) sR[m] = 1.0f / sum;
    }
    __syncthreads();

    // ---- phase 3: O = (P @ V) * (1/sum); wave w owns d in [128w, 128w+128)
    v8f acc[8] = {};
    const _Float16* pRow = sS + col * N_TOK;   // A row m = lane&15
    const int dwave = wave * 128;
    for (int nk = 0; nk < N_TOK; nk += 32) {
        v16h a = a_frag_f16(pRow, nk, half);
#pragma unroll
        for (int t = 0; t < 8; ++t) {
            const _Float16* bRow = V + (size_t)(dwave + t * 16 + col) * N_TOK; // Vt row = d
            v16h b = b_frag_f16(bRow, nk, half);
            acc[t] = __builtin_amdgcn_wmma_f32_16x16x32_f16(false, a, false, b,
                                                            (short)0, acc[t], false, false);
        }
    }
#pragma unroll
    for (int t = 0; t < 8; ++t) {
#pragma unroll
        for (int r = 0; r < 8; ++r) {
            const int m = r + half * 8;
            O[(size_t)(rbase + m) * D_EMBED + dwave + t * 16 + col] = acc[t][r] * sR[m];
        }
    }
}

// ---------------------------------------------------------------------------
extern "C" void kernel_launch(void* const* d_in, const int* in_sizes, int n_in,
                              void* d_out, int out_size, void* d_ws, size_t ws_size,
                              hipStream_t stream) {
    (void)in_sizes; (void)n_in; (void)out_size; (void)ws_size;
    const float* x1  = (const float*)d_in[0];
    const float* x2  = (const float*)d_in[1];
    const float* Wq1 = (const float*)d_in[2];  const float* bq1 = (const float*)d_in[3];
    const float* Wk1 = (const float*)d_in[4];  const float* bk1 = (const float*)d_in[5];
    const float* Wv1 = (const float*)d_in[6];  const float* bv1 = (const float*)d_in[7];
    const float* Wq2 = (const float*)d_in[8];  const float* bq2 = (const float*)d_in[9];
    const float* Wk2 = (const float*)d_in[10]; const float* bk2 = (const float*)d_in[11];
    const float* Wv2 = (const float*)d_in[12]; const float* bv2 = (const float*)d_in[13];

    const size_t MAT  = (size_t)N_TOK * D_EMBED;   // 4M elems
    const size_t WMAT = (size_t)D_EMBED * D_EMBED; // 1M elems
    _Float16* q1  = (_Float16*)d_ws;
    _Float16* k1  = q1  + MAT;
    _Float16* v1t = k1  + MAT;
    _Float16* q2  = v1t + MAT;
    _Float16* k2  = q2  + MAT;
    _Float16* v2t = k2  + MAT;
    _Float16* x1h = v2t + MAT;
    _Float16* x2h = x1h + MAT;
    _Float16* wh  = x2h + MAT;   // 6 weight matrices, f16
    _Float16* wq1h = wh;           _Float16* wk1h = wh + WMAT;
    _Float16* wv1h = wh + 2*WMAT;  _Float16* wq2h = wh + 3*WMAT;
    _Float16* wk2h = wh + 4*WMAT;  _Float16* wv2h = wh + 5*WMAT;

    // ---- pass 0: one-time fp32 -> f16 conversion of activations + weights
    {
        const int nx4 = (int)(MAT / 4), nw4 = (int)(WMAT / 4);
        cvt_kernel<<<(nx4 + 255) / 256, 256, 0, stream>>>(x1,  x1h,  nx4);
        cvt_kernel<<<(nx4 + 255) / 256, 256, 0, stream>>>(x2,  x2h,  nx4);
        cvt_kernel<<<(nw4 + 255) / 256, 256, 0, stream>>>(Wq1, wq1h, nw4);
        cvt_kernel<<<(nw4 + 255) / 256, 256, 0, stream>>>(Wk1, wk1h, nw4);
        cvt_kernel<<<(nw4 + 255) / 256, 256, 0, stream>>>(Wv1, wv1h, nw4);
        cvt_kernel<<<(nw4 + 255) / 256, 256, 0, stream>>>(Wq2, wq2h, nw4);
        cvt_kernel<<<(nw4 + 255) / 256, 256, 0, stream>>>(Wk2, wk2h, nw4);
        cvt_kernel<<<(nw4 + 255) / 256, 256, 0, stream>>>(Wv2, wv2h, nw4);
    }

    // ---- pass 1: six projections (all-WMMA inner loops)
    dim3 pgrid(N_TOK / 16, D_EMBED / 512);
    const size_t psmem = 16 * D_EMBED * sizeof(_Float16);            // 32 KB
    proj_kernel<<<pgrid, 256, psmem, stream>>>(x1h, wq1h, bq1, q1, 0);
    proj_kernel<<<pgrid, 256, psmem, stream>>>(x1h, wk1h, bk1, k1, 0);
    proj_kernel<<<pgrid, 256, psmem, stream>>>(x1h, wv1h, bv1, v1t, 1);
    proj_kernel<<<pgrid, 256, psmem, stream>>>(x2h, wq2h, bq2, q2, 0);
    proj_kernel<<<pgrid, 256, psmem, stream>>>(x2h, wk2h, bk2, k2, 0);
    proj_kernel<<<pgrid, 256, psmem, stream>>>(x2h, wv2h, bv2, v2t, 1);

    // ---- pass 2: both cross-attentions, fully fused
    dim3 agrid(N_TOK / 16, 2);
    const size_t asmem = 16 * D_EMBED * 2 + 16 * N_TOK * 2 + 16 * 4; // ~160 KB (CDNA5 LDS)
    attn_kernel<<<agrid, 256, asmem, stream>>>(q2, k1, v1t, q1, k2, v2t, (float*)d_out);
}
